// My_legacyLSTM_40913858462325
// MI455X (gfx1250) — compile-verified, matrix operand order checked by reading, not run on hardware
//
#include <hip/hip_runtime.h>
#include <hip/hip_bf16.h>

// ---------------------------------------------------------------------------
// LSTM on MI455X (gfx1250): bf16 WMMA recurrence, f32 accumulation,
// async global->LDS staging (ASYNCcnt path) on the sequential critical path.
// ---------------------------------------------------------------------------

typedef unsigned short ushort_t;
typedef __attribute__((ext_vector_type(16))) __bf16 v16bf;
typedef __attribute__((ext_vector_type(8)))  float  v8f;

#define BATCH  64
#define SEQ    512
#define INSZ   1024
#define HID    1024
#define KTOT   2048   // INSZ + HID
#define NGATE  4

__device__ __forceinline__ ushort_t f32_to_bf16_rne(float f) {
  union { float f; unsigned int u; } v; v.f = f;
  unsigned int r = v.u + 0x7fffu + ((v.u >> 16) & 1u);
  return (ushort_t)(r >> 16);
}

// Async 16B global -> LDS copy (CDNA5, tracked by ASYNCcnt).
// GVS addressing: mem = SGPR64(base) + VGPR32(byte offset).
__device__ __forceinline__ void async_b128_to_lds(unsigned lds_byte_off,
                                                  const ushort_t* base,
                                                  unsigned byte_off) {
  asm volatile("global_load_async_to_lds_b128 %0, %1, %2"
               :
               : "v"(lds_byte_off), "v"(byte_off), "s"(base)
               : "memory");
}

__device__ __forceinline__ void wait_async0() {
  asm volatile("s_wait_asynccnt 0" ::: "memory");
}

// ---------------------------------------------------------------------------
// Pack [W_g ; U_g] (f32, row-major [k][n]) into bf16 column-major [g][n][k].
// 32x32 LDS tile transpose; block = (32,8), grid = (KTOT/32, HID/32, 4).
// ---------------------------------------------------------------------------
__global__ __launch_bounds__(256) void pack_weights_kernel(
    const float* __restrict__ wf, const float* __restrict__ uf,
    const float* __restrict__ wi, const float* __restrict__ ui,
    const float* __restrict__ wo, const float* __restrict__ uo,
    const float* __restrict__ wc, const float* __restrict__ uc,
    ushort_t* __restrict__ wcat)
{
  __shared__ float tile[32][33];
  const int g = blockIdx.z;
  const float* wsrc = (g == 0) ? wf : (g == 1) ? wi : (g == 2) ? wo : wc;
  const float* usrc = (g == 0) ? uf : (g == 1) ? ui : (g == 2) ? uo : uc;
  const int k0 = blockIdx.x * 32;   // 0..2047
  const int n0 = blockIdx.y * 32;   // 0..1023
  const int tx = threadIdx.x, ty = threadIdx.y;

  #pragma unroll
  for (int i = 0; i < 32; i += 8) {
    int k = k0 + ty + i;
    const float* src = (k < INSZ) ? (wsrc + (size_t)k * HID)
                                  : (usrc + (size_t)(k - INSZ) * HID);
    tile[ty + i][tx] = src[n0 + tx];
  }
  __syncthreads();
  #pragma unroll
  for (int i = 0; i < 32; i += 8) {
    int n = n0 + ty + i;
    int k = k0 + tx;
    wcat[((size_t)g * HID + n) * KTOT + k] = f32_to_bf16_rne(tile[tx][ty + i]);
  }
}

// ---------------------------------------------------------------------------
// Convert the whole x tensor to bf16 once (removes per-step conversion work
// from the 512-step sequential critical path).  8 elems / thread.
// ---------------------------------------------------------------------------
__global__ __launch_bounds__(256) void convert_x_kernel(
    const float* __restrict__ x, ushort_t* __restrict__ xbf)
{
  size_t i = ((size_t)blockIdx.x * 256 + threadIdx.x) * 8;
  float4 a = *(const float4*)(x + i);
  float4 b = *(const float4*)(x + i + 4);
  uint4 p;
  p.x = (unsigned)f32_to_bf16_rne(a.x) | ((unsigned)f32_to_bf16_rne(a.y) << 16);
  p.y = (unsigned)f32_to_bf16_rne(a.z) | ((unsigned)f32_to_bf16_rne(a.w) << 16);
  p.z = (unsigned)f32_to_bf16_rne(b.x) | ((unsigned)f32_to_bf16_rne(b.y) << 16);
  p.w = (unsigned)f32_to_bf16_rne(b.z) | ((unsigned)f32_to_bf16_rne(b.w) << 16);
  *(uint4*)(xbf + i) = p;
}

// ---------------------------------------------------------------------------
// Zero h ping-pong (bf16) and c (f32).  grid*block covers 2*BATCH*HID.
// ---------------------------------------------------------------------------
__global__ __launch_bounds__(256) void init_state_kernel(ushort_t* __restrict__ h,
                                                         float* __restrict__ c)
{
  int i = blockIdx.x * 256 + threadIdx.x;          // i < 2*BATCH*HID
  h[i] = (ushort_t)0;
  if (i < BATCH * HID) c[i] = 0.0f;
}

// ---------------------------------------------------------------------------
// One LSTM timestep.  grid = (HID/16, BATCH/16), block = 128 (4 waves).
// Wave g computes gate g's 16x16 pre-activation tile via 64 bf16 WMMAs
// (K = 2048 = x-part 1024 + h-part 1024, async-staged through LDS).
// ---------------------------------------------------------------------------
__global__ __launch_bounds__(128) void lstm_step_kernel(
    const ushort_t* __restrict__ xbf,     // [BATCH][SEQ][INSZ] bf16
    const ushort_t* __restrict__ wcat,    // [4][HID][KTOT] bf16
    const float*    __restrict__ b_f, const float* __restrict__ b_i,
    const float*    __restrict__ b_o, const float* __restrict__ b_c,
    const ushort_t* __restrict__ h_prev,  // [BATCH][HID] bf16
    ushort_t*       __restrict__ h_next,  // [BATCH][HID] bf16
    float*          __restrict__ c_state, // [BATCH][HID]
    float*          __restrict__ ys,      // [BATCH][SEQ][HID]
    int t)
{
  __shared__ __align__(64) ushort_t A_lds[16][1024];      // 32 KiB
  __shared__ float gates_lds[NGATE][16][16];              //  4 KiB

  const int tid  = threadIdx.x;
  const int lane = tid & 31;
  const int gate = tid >> 5;
  const int nt   = blockIdx.x;          // hidden-column group
  const int mt   = blockIdx.y;          // batch-row group

  // B-fragment base: lane holds column (nt*16 + lane%16), K from lane/16*16.
  const int ncol = nt * 16 + (lane & 15);
  const ushort_t* bbase = wcat + ((size_t)gate * HID + ncol) * KTOT
                               + ((lane >> 4) << 4);
  // A-fragment: lane holds row lane%16, K from lane/16*16.
  const int arow = lane & 15;
  const int koff = (lane >> 4) << 4;

  // LDS byte offset of A_lds (generic LDS address carries offset in [31:0]).
  const unsigned lds_base = (unsigned)(uintptr_t)&A_lds[0][0];
  const unsigned lds_thr  = lds_base + (unsigned)tid * 16u;   // 8 ushorts

  v8f acc = {};

  #pragma unroll
  for (int phase = 0; phase < 2; ++phase) {
    __syncthreads();   // prior-phase readers done before re-staging A_lds

    // Stage one 16x1024 bf16 A panel: 16 rounds x 128 lanes x 16B, all async.
    // Round 'it' copies row 'it'; thread tid covers columns tid*8 .. tid*8+7.
    if (phase == 0) {
      #pragma unroll
      for (int it = 0; it < 16; ++it) {
        unsigned src_off =
            (((unsigned)(mt * 16 + it) * SEQ + (unsigned)t) * INSZ +
             (unsigned)tid * 8u) * 2u;
        async_b128_to_lds(lds_thr + (unsigned)it * 2048u, xbf, src_off);
      }
    } else {
      #pragma unroll
      for (int it = 0; it < 16; ++it) {
        unsigned src_off =
            ((unsigned)(mt * 16 + it) * HID + (unsigned)tid * 8u) * 2u;
        async_b128_to_lds(lds_thr + (unsigned)it * 2048u, h_prev, src_off);
      }
    }
    wait_async0();
    __syncthreads();

    const ushort_t* bptr = bbase + phase * INSZ;
    #pragma unroll 4
    for (int kk = 0; kk < 1024; kk += 32) {
      v16bf amat = *(const v16bf*)&A_lds[arow][kk + koff];
      v16bf bmat = *(const v16bf*)(bptr + kk);
      acc = __builtin_amdgcn_wmma_f32_16x16x32_bf16(
          /*neg_a=*/false, amat, /*neg_b=*/false, bmat,
          /*c_mod=*/(short)0, acc, /*reuse_a=*/false, /*reuse_b=*/false);
    }
  }

  // Spill this wave's 16x16 f32 tile (C layout: vgpr v -> row v + (lane/16)*8).
  #pragma unroll
  for (int v = 0; v < 8; ++v) {
    int m = v + ((lane >> 4) << 3);
    gates_lds[gate][m][lane & 15] = acc[v];
  }
  __syncthreads();

  // Cell update: 256 outputs, 2 per thread.
  for (int e = tid; e < 256; e += 128) {
    int m = e >> 4, n = e & 15;
    int b = mt * 16 + m;
    int j = nt * 16 + n;
    float pf = gates_lds[0][m][n] + b_f[j];
    float pi = gates_lds[1][m][n] + b_i[j];
    float po = gates_lds[2][m][n] + b_o[j];
    float pc = gates_lds[3][m][n] + b_c[j];
    float fg = 1.0f / (1.0f + __expf(-pf));
    float ig = 1.0f / (1.0f + __expf(-pi));
    float og = 1.0f / (1.0f + __expf(-po));
    float gg = tanhf(pc);
    size_t ci = (size_t)b * HID + j;
    float cn = fg * c_state[ci] + ig * gg;
    c_state[ci] = cn;
    float h = og * tanhf(cn);
    ys[((size_t)b * SEQ + t) * HID + j] = h;
    h_next[ci] = f32_to_bf16_rne(h);
  }
}

// ---------------------------------------------------------------------------
// Copy final (h, c) to the output tail.  Covers BATCH*HID elements.
// ---------------------------------------------------------------------------
__global__ __launch_bounds__(256) void finalize_kernel(
    const float* __restrict__ ys, const float* __restrict__ c_state,
    float* __restrict__ tail)
{
  int i = blockIdx.x * 256 + threadIdx.x;          // i < BATCH*HID
  int b = i >> 10, j = i & 1023;
  tail[i] = ys[((size_t)b * SEQ + (SEQ - 1)) * HID + j];  // final h
  tail[BATCH * HID + i] = c_state[i];                     // final c
}

// ---------------------------------------------------------------------------
extern "C" void kernel_launch(void* const* d_in, const int* in_sizes, int n_in,
                              void* d_out, int out_size, void* d_ws, size_t ws_size,
                              hipStream_t stream) {
  (void)in_sizes; (void)n_in; (void)out_size; (void)ws_size;

  const float* x   = (const float*)d_in[0];
  const float* w_f = (const float*)d_in[1];
  const float* u_f = (const float*)d_in[2];
  const float* b_f = (const float*)d_in[3];
  const float* w_i = (const float*)d_in[4];
  const float* u_i = (const float*)d_in[5];
  const float* b_i = (const float*)d_in[6];
  const float* w_o = (const float*)d_in[7];
  const float* u_o = (const float*)d_in[8];
  const float* b_o = (const float*)d_in[9];
  const float* w_c = (const float*)d_in[10];
  const float* u_c = (const float*)d_in[11];
  const float* b_c = (const float*)d_in[12];

  float* out = (float*)d_out;
  char*  ws  = (char*)d_ws;

  const size_t WCAT_BYTES = (size_t)NGATE * HID * KTOT * sizeof(ushort_t); // 16 MiB
  const size_t XBF_BYTES  = (size_t)BATCH * SEQ * INSZ * sizeof(ushort_t); // 64 MiB
  const size_t HBUF_BYTES = (size_t)2 * BATCH * HID * sizeof(ushort_t);    // 256 KiB
  ushort_t* wcat = (ushort_t*)ws;
  ushort_t* xbf  = (ushort_t*)(ws + WCAT_BYTES);
  ushort_t* hbuf = (ushort_t*)(ws + WCAT_BYTES + XBF_BYTES);
  float*    cbuf = (float*)(ws + WCAT_BYTES + XBF_BYTES + HBUF_BYTES);
  float*    ys   = out;

  // 1) Pack weights into bf16 [gate][n][k].
  pack_weights_kernel<<<dim3(KTOT / 32, HID / 32, NGATE), dim3(32, 8), 0, stream>>>(
      w_f, u_f, w_i, u_i, w_o, u_o, w_c, u_c, wcat);

  // 2) Convert x to bf16 once (off the sequential critical path).
  convert_x_kernel<<<(BATCH * SEQ * INSZ) / (256 * 8), 256, 0, stream>>>(x, xbf);

  // 3) Zero h (both parities) and c.
  init_state_kernel<<<(2 * BATCH * HID) / 256, 256, 0, stream>>>(hbuf, cbuf);

  // 4) 512 sequential timesteps (stream-ordered; graph-capture friendly).
  for (int t = 0; t < SEQ; ++t) {
    ushort_t* hp = hbuf + (size_t)(t & 1) * BATCH * HID;
    ushort_t* hn = hbuf + (size_t)((t + 1) & 1) * BATCH * HID;
    lstm_step_kernel<<<dim3(HID / 16, BATCH / 16), 128, 0, stream>>>(
        xbf, wcat, b_f, b_i, b_o, b_c, hp, hn, cbuf, ys, t);
  }

  // 5) Final (h, c) tuple outputs.
  finalize_kernel<<<(BATCH * HID) / 256, 256, 0, stream>>>(
      ys, cbuf, out + (size_t)BATCH * SEQ * HID);
}
